// Attention_91259465105689
// MI455X (gfx1250) — matrix operand order repaired and checked by reading, not compiled, not run on hardware
//
#include <hip/hip_runtime.h>
#include <math.h>

// ---------------------------------------------------------------------------
// Fused attention layer for MI455X (gfx1250, wave32, WMMA bf16).
//   x(B,S,D) -> Q/K/V proj -> per-head RMSNorm+RoPE(+gain) -> causal GQA
//   flash-attention -> output projection.
// Matmuls: V_WMMA_F32_16X16X32_BF16. Attention K-tiles staged with
// GLOBAL_LOAD_ASYNC_TO_LDS_B128 (ASYNCcnt) + s_wait_asynccnt.
// ---------------------------------------------------------------------------

typedef unsigned short u16;
typedef __attribute__((ext_vector_type(16))) __bf16    v16bf;
typedef __attribute__((ext_vector_type(8)))  float     v8f;
typedef __attribute__((ext_vector_type(4)))  float     v4f;
typedef __attribute__((ext_vector_type(4)))  unsigned  v4u;
typedef __attribute__((ext_vector_type(2)))  unsigned  v2u;

#define B_    2
#define S_    2048
#define D_    1024
#define H_    16
#define KVH_  4
#define HD_   64
#define M_    (B_ * S_)      // 4096 rows
#define KVD_  (KVH_ * HD_)   // 256

static __device__ __forceinline__ u16 f2bf(float f) {
  unsigned u = __float_as_uint(f);
  u += 0x7fffu + ((u >> 16) & 1u);   // round-to-nearest-even
  return (u16)(u >> 16);
}

union FragBF { v4u u[2]; v16bf v; }; // 32B: one 16x32 A-frag or 32x16 B-frag

static __device__ __forceinline__ v8f zero8() {
  v8f z = {0.f, 0.f, 0.f, 0.f, 0.f, 0.f, 0.f, 0.f};
  return z;
}

static __device__ __forceinline__ v8f wmma_bf16(v16bf a, v16bf b, v8f c) {
  // D = A(16x32) * B(32x16) + C, f32 accumulate
  return __builtin_amdgcn_wmma_f32_16x16x32_bf16(
      /*neg_a=*/false, a, /*neg_b=*/false, b,
      /*c_mod=*/(short)0, c, /*reuse_a=*/false, /*reuse_b=*/false);
}

// ---------------------------------------------------------------------------
// GEMM: C[M,N] = A[M,K] * W[N,K]^T   (both operands K-contiguous, f32 in,
// converted to bf16 while staging to LDS, f32 out). 128x128 tile, 8 waves.
// N,K compile-time so row strides fold into instruction offsets.
// ---------------------------------------------------------------------------
template<int N, int K>
__global__ __launch_bounds__(256) void gemm_xwT(
    const float* __restrict__ A, const float* __restrict__ W,
    float* __restrict__ C)
{
  __shared__ u16 At[128 * 40];   // 32 bf16 per row, padded to 40 (80B stride)
  __shared__ u16 Bt[128 * 40];

  const int tid   = threadIdx.x;
  const int lane  = tid & 31;
  const int wid   = tid >> 5;
  const int wm    = (wid & 3) * 32;   // wave row offset  (4 wave-rows)
  const int wn    = (wid >> 2) * 64;  // wave col offset  (2 wave-cols)
  const int mBase = blockIdx.y * 128;
  const int nBase = blockIdx.x * 128;

  v8f acc[2][4];
  #pragma unroll
  for (int i = 0; i < 2; i++)
    #pragma unroll
    for (int j = 0; j < 4; j++) acc[i][j] = zero8();

  for (int k0 = 0; k0 < K; k0 += 32) {
    // ---- stage 128x32 A and W tiles, f32 -> bf16 ----
    #pragma unroll
    for (int i = 0; i < 4; i++) {
      int g   = tid * 4 + i;        // 0..1023 float4 chunks
      int row = g >> 3;             // 0..127
      int col = (g & 7) * 4;        // 0..28
      v4f av = *(const v4f*)&A[(size_t)(mBase + row) * K + k0 + col];
      v4f wv = *(const v4f*)&W[(size_t)(nBase + row) * K + k0 + col];
      u16 ap[4] = { f2bf(av.x), f2bf(av.y), f2bf(av.z), f2bf(av.w) };
      u16 wp[4] = { f2bf(wv.x), f2bf(wv.y), f2bf(wv.z), f2bf(wv.w) };
      *(v2u*)&At[row * 40 + col] = *(const v2u*)ap;
      *(v2u*)&Bt[row * 40 + col] = *(const v2u*)wp;
    }
    __syncthreads();

    // ---- fragment loads (wave32 WMMA layouts) ----
    const int khalfA = (lane & 16) ? 16 : 0;  // A: lanes16-31 hold K=16..31
    const int kstB   = (lane & 16) ? 16 : 0;  // B: lanes16-31 hold K=16..31
    FragBF a[2], b[4];
    #pragma unroll
    for (int im = 0; im < 2; im++) {
      const u16* p = &At[(wm + im * 16 + (lane & 15)) * 40 + khalfA];
      a[im].u[0] = *(const v4u*)p;
      a[im].u[1] = *(const v4u*)(p + 8);
    }
    #pragma unroll
    for (int jn = 0; jn < 4; jn++) {
      const u16* p = &Bt[(wn + jn * 16 + (lane & 15)) * 40 + kstB];
      b[jn].u[0] = *(const v4u*)p;
      b[jn].u[1] = *(const v4u*)(p + 8);
    }
    #pragma unroll
    for (int im = 0; im < 2; im++)
      #pragma unroll
      for (int jn = 0; jn < 4; jn++)
        acc[im][jn] = wmma_bf16(a[im].v, b[jn].v, acc[im][jn]);
    __syncthreads();
  }

  const int rsel = (lane & 16) ? 8 : 0;  // C layout: lanes16-31 hold rows 8..15
  float* cbase = C + (size_t)(mBase + wm + rsel) * N + nBase + wn + (lane & 15);
  #pragma unroll
  for (int im = 0; im < 2; im++)
    #pragma unroll
    for (int jn = 0; jn < 4; jn++)
      #pragma unroll
      for (int r = 0; r < 8; r++)
        cbase[(size_t)(im * 16 + r) * N + jn * 16] = acc[im][jn][r];
}

// ---------------------------------------------------------------------------
// Per-head RMSNorm + RoPE (+ q_gain), f32 -> bf16 in [B][H][S][64] layout.
// One wave per head-task; lane l owns the RoPE pair (dim l, dim l+32).
// ---------------------------------------------------------------------------
__global__ __launch_bounds__(256) void rmsrope(
    const float* __restrict__ qf, const float* __restrict__ kf,
    const float* __restrict__ vf, const float* __restrict__ qgain,
    u16* __restrict__ qb, u16* __restrict__ kb, u16* __restrict__ vb)
{
  const int s    = blockIdx.x;
  const int b    = blockIdx.y;
  const int lane = threadIdx.x & 31;
  const int wid  = threadIdx.x >> 5;

  const float inv_freq = __powf(10000.f, -(float)lane * (1.f / 32.f));
  const float ang = (float)s * inv_freq;
  float sn, cs;
  __sincosf(ang, &sn, &cs);

  for (int task = wid; task < 24; task += 8) {
    if (task < 16) {                      // Q heads: norm + rope + gain
      int h = task;
      const float* src = qf + ((size_t)(b * S_ + s)) * D_ + h * HD_;
      float x1 = src[lane], x2 = src[lane + 32];
      float ss = x1 * x1 + x2 * x2;
      #pragma unroll
      for (int off = 16; off; off >>= 1) ss += __shfl_xor(ss, off, 32);
      float sc = rsqrtf(ss * (1.f / 64.f) + 1.1920929e-7f) * qgain[h];
      float y1 = x1 * sc, y2 = x2 * sc;
      u16* dst = qb + ((size_t)((b * H_ + h) * S_) + s) * HD_;
      dst[lane]      = f2bf( y1 * cs + y2 * sn);
      dst[lane + 32] = f2bf(-y1 * sn + y2 * cs);
    } else if (task < 20) {               // K heads: norm + rope
      int h = task - 16;
      const float* src = kf + ((size_t)(b * S_ + s)) * KVD_ + h * HD_;
      float x1 = src[lane], x2 = src[lane + 32];
      float ss = x1 * x1 + x2 * x2;
      #pragma unroll
      for (int off = 16; off; off >>= 1) ss += __shfl_xor(ss, off, 32);
      float sc = rsqrtf(ss * (1.f / 64.f) + 1.1920929e-7f);
      float y1 = x1 * sc, y2 = x2 * sc;
      u16* dst = kb + ((size_t)((b * KVH_ + h) * S_) + s) * HD_;
      dst[lane]      = f2bf( y1 * cs + y2 * sn);
      dst[lane + 32] = f2bf(-y1 * sn + y2 * cs);
    } else {                              // V heads: convert only
      int h = task - 20;
      const float* src = vf + ((size_t)(b * S_ + s)) * KVD_ + h * HD_;
      u16* dst = vb + ((size_t)((b * KVH_ + h) * S_) + s) * HD_;
      dst[lane]      = f2bf(src[lane]);
      dst[lane + 32] = f2bf(src[lane + 32]);
    }
  }
}

// ---------------------------------------------------------------------------
// Flash attention: block = (64 query rows, head, batch); 4 waves x 16 rows.
// All waves share the same causal key-block count, so K / V^T tiles are
// staged cooperatively once per 64-key block. K tile uses the CDNA5 async
// DMA path (global_load_async_to_lds_b128 / ASYNCcnt); V is transposed
// through registers. Only the diagonal block masks.
// ---------------------------------------------------------------------------
__global__ __launch_bounds__(128) void attn(
    const u16* __restrict__ qb, const u16* __restrict__ kbm,
    const u16* __restrict__ vbm, float* __restrict__ ctx)
{
  __shared__ u16 Kt[64 * 72];       // [key][dim], padded stride 72
  __shared__ u16 Vt[64 * 72];       // [dim][key] (transposed), padded
  __shared__ u16 Pt[4][16 * 72];    // per-wave P tile (C-layout -> A-layout)

  const int qtile = blockIdx.x;
  const int h     = blockIdx.y;
  const int b     = blockIdx.z;
  const int hk    = h >> 2;         // GQA: 4 query heads per kv head
  const int tid   = threadIdx.x;
  const int lane  = tid & 31;
  const int wid   = tid >> 5;
  const int qr0   = qtile * 64 + wid * 16;
  const int rsel  = (lane & 16) ? 8 : 0;
  const int khalf = (lane & 16) ? 16 : 0;
  const int kst16 = (lane & 16) ? 16 : 0;

  // Q A-fragments (head dim 64 = two K-steps of 32), loaded once.
  FragBF aq[2];
  {
    const u16* qp = qb + ((size_t)((b * H_ + h) * S_) + qr0 + (lane & 15)) * HD_;
    #pragma unroll
    for (int t = 0; t < 2; t++) {
      aq[t].u[0] = *(const v4u*)(qp + t * 32 + khalf);
      aq[t].u[1] = *(const v4u*)(qp + t * 32 + khalf + 8);
    }
  }

  float mr[8], lr[8];
  #pragma unroll
  for (int r = 0; r < 8; r++) { mr[r] = -3.0e38f; lr[r] = 0.f; }
  v8f o[4];
  #pragma unroll
  for (int j = 0; j < 4; j++) o[j] = zero8();

  const u16* kg0 = kbm + ((size_t)((b * KVH_ + hk) * S_)) * HD_;
  const u16* vg0 = vbm + ((size_t)((b * KVH_ + hk) * S_)) * HD_;

  for (int kbk = 0; kbk <= qtile; kbk++) {
    const u16* kg = kg0 + (size_t)kbk * 64 * HD_;
    const u16* vg = vg0 + (size_t)kbk * 64 * HD_;

    // ---- K tile: hardware async DMA global -> LDS (no VGPR round trip) ----
    {
      int g = tid;                         // 128 lanes x 16B x 4 reps = 8KB
      #pragma unroll
      for (int rep = 0; rep < 4; rep++, g += 128) {
        int row = g >> 3, seg = (g & 7) * 8;
        unsigned    ldsa = (unsigned)(uintptr_t)&Kt[row * 72 + seg];
        const u16*  gptr = kg + row * 64 + seg;
        asm volatile("global_load_async_to_lds_b128 %0, %1, off"
                     :: "v"(ldsa), "v"(gptr) : "memory");
      }
    }
    // ---- V tile: load rows, scatter transposed into LDS ----
    for (int g = tid; g < 512; g += 128) {
      int row = g >> 3, seg = (g & 7) * 8;
      union { v4u q; u16 s[8]; } vv;
      vv.q = *(const v4u*)(vg + row * 64 + seg);
      #pragma unroll
      for (int j = 0; j < 8; j++) Vt[(seg + j) * 72 + row] = vv.s[j];
    }
    asm volatile("s_wait_asynccnt 0" ::: "memory");
    __syncthreads();

    // prefetch next key block while computing on this one
    if (kbk < qtile) {
      __builtin_prefetch((const char*)(kg + 64 * HD_) + tid * 64, 0, 1);
      __builtin_prefetch((const char*)(vg + 64 * HD_) + tid * 64, 0, 1);
    }

    // ---- S = Q K^T (scaled) ----
    v8f sf[4];
    #pragma unroll
    for (int nf = 0; nf < 4; nf++) {
      sf[nf] = zero8();
      #pragma unroll
      for (int t = 0; t < 2; t++) {
        FragBF bk;
        const u16* p = &Kt[(nf * 16 + (lane & 15)) * 72 + t * 32 + kst16];
        bk.u[0] = *(const v4u*)p;
        bk.u[1] = *(const v4u*)(p + 8);
        sf[nf] = wmma_bf16(aq[t].v, bk.v, sf[nf]);
      }
    }

    // ---- online softmax (rows live in lane groups of 16) ----
    const bool maskblk = (kbk == qtile);
    #pragma unroll
    for (int r = 0; r < 8; r++) {
      int m = qr0 + rsel + r;
      float best = -3.0e38f;
      #pragma unroll
      for (int nf = 0; nf < 4; nf++) {
        float v = sf[nf][r] * 0.125f;            // 1/sqrt(64)
        if (maskblk) {
          int n = kbk * 64 + nf * 16 + (lane & 15);
          if (n > m) v = -3.0e38f;
        }
        sf[nf][r] = v;
        best = fmaxf(best, v);
      }
      #pragma unroll
      for (int off = 1; off < 16; off <<= 1)
        best = fmaxf(best, __shfl_xor(best, off, 16));
      float newm  = fmaxf(mr[r], best);
      float alpha = __expf(mr[r] - newm);
      mr[r] = newm;
      float sum = 0.f;
      #pragma unroll
      for (int nf = 0; nf < 4; nf++) {
        float pv = __expf(sf[nf][r] - newm);
        sf[nf][r] = pv;
        sum += pv;
      }
      #pragma unroll
      for (int off = 1; off < 16; off <<= 1) sum += __shfl_xor(sum, off, 16);
      lr[r] = lr[r] * alpha + sum;
      #pragma unroll
      for (int j = 0; j < 4; j++) o[j][r] *= alpha;
    }

    // ---- P: C-layout -> A-layout via per-wave LDS tile ----
    #pragma unroll
    for (int nf = 0; nf < 4; nf++)
      #pragma unroll
      for (int r = 0; r < 8; r++)
        Pt[wid][(rsel + r) * 72 + nf * 16 + (lane & 15)] = f2bf(sf[nf][r]);
    asm volatile("s_wait_dscnt 0" ::: "memory");  // intra-wave LDS RAW fence
    FragBF pa[2];
    {
      const u16* pp = &Pt[wid][(lane & 15) * 72];
      #pragma unroll
      for (int t = 0; t < 2; t++) {
        pa[t].u[0] = *(const v4u*)(pp + t * 32 + khalf);
        pa[t].u[1] = *(const v4u*)(pp + t * 32 + khalf + 8);
      }
    }

    // ---- O += P V (B-frags from transposed V tile, contiguous runs) ----
    #pragma unroll
    for (int t = 0; t < 2; t++)
      #pragma unroll
      for (int nf = 0; nf < 4; nf++) {
        FragBF bv;
        const u16* p = &Vt[(nf * 16 + (lane & 15)) * 72 + t * 32 + kst16];
        bv.u[0] = *(const v4u*)p;
        bv.u[1] = *(const v4u*)(p + 8);
        o[nf] = wmma_bf16(pa[t].v, bv.v, o[nf]);
      }
    __syncthreads();
  }

  // ---- normalize and write ctx in (B,S,D) layout for the O-projection ----
  float* cb = ctx + ((size_t)(b * S_ + qr0 + rsel)) * D_ + h * HD_ + (lane & 15);
  #pragma unroll
  for (int nf = 0; nf < 4; nf++)
    #pragma unroll
    for (int r = 0; r < 8; r++)
      cb[(size_t)r * D_ + nf * 16] = o[nf][r] / lr[r];
}

// ---------------------------------------------------------------------------
extern "C" void kernel_launch(void* const* d_in, const int* in_sizes, int n_in,
                              void* d_out, int out_size, void* d_ws, size_t ws_size,
                              hipStream_t stream)
{
  const float* x  = (const float*)d_in[0];
  const float* wq = (const float*)d_in[1];
  const float* wk = (const float*)d_in[2];
  const float* wv = (const float*)d_in[3];
  const float* wo = (const float*)d_in[4];
  const float* qg = (const float*)d_in[5];
  float* out = (float*)d_out;

  // Workspace layout (36 MB total): ctx aliases qf (consumed before reuse).
  char* w = (char*)d_ws;
  float* qf  = (float*)w;  w += (size_t)M_ * D_   * 4;  // 16 MB (reused as ctx)
  float* kf  = (float*)w;  w += (size_t)M_ * KVD_ * 4;  //  4 MB
  float* vf  = (float*)w;  w += (size_t)M_ * KVD_ * 4;  //  4 MB
  u16*   qbm = (u16*)w;    w += (size_t)M_ * D_   * 2;  //  8 MB
  u16*   kbm = (u16*)w;    w += (size_t)M_ * KVD_ * 2;  //  2 MB
  u16*   vbm = (u16*)w;    w += (size_t)M_ * KVD_ * 2;  //  2 MB
  float* ctx = qf;

  dim3 blk(256);
  gemm_xwT<D_,   D_><<<dim3(D_   / 128, M_ / 128), blk, 0, stream>>>(x, wq, qf);
  gemm_xwT<KVD_, D_><<<dim3(KVD_ / 128, M_ / 128), blk, 0, stream>>>(x, wk, kf);
  gemm_xwT<KVD_, D_><<<dim3(KVD_ / 128, M_ / 128), blk, 0, stream>>>(x, wv, vf);
  rmsrope<<<dim3(S_, B_), blk, 0, stream>>>(qf, kf, vf, qg, qbm, kbm, vbm);
  attn<<<dim3(S_ / 64, H_, B_), dim3(128), 0, stream>>>(qbm, kbm, vbm, ctx);
  gemm_xwT<D_, D_><<<dim3(D_ / 128, M_ / 128), blk, 0, stream>>>(ctx, wo, out);
}
